// NodeGraphTransformerLayer_68461778698585
// MI455X (gfx1250) — matrix operand order repaired and checked by reading, not compiled, not run on hardware
//
#include <hip/hip_runtime.h>
#include <hip/hip_bf16.h>
#include <math.h>

#define NN 50000
#define EE 800000
#define DIN 256
#define DOUT 256
#define HH 8
#define HD 32

typedef __attribute__((ext_vector_type(16))) __bf16 bf16x16;
typedef __attribute__((ext_vector_type(8)))  float  floatx8;

// ---------------------------------------------------------------------------
// elementwise f32 -> bf16
__global__ void k_f32_to_bf16(const float* __restrict__ src, __bf16* __restrict__ dst, int n) {
    int i = blockIdx.x * 256 + threadIdx.x;
    if (i < n) dst[i] = (__bf16)src[i];
}

// transpose + convert: W (fi x fo, f32, row-major) -> WT (fo x fi, bf16)
__global__ void k_transpose_bf16(const float* __restrict__ W, __bf16* __restrict__ WT,
                                 int fi, int fo) {
    int i = blockIdx.x * 256 + threadIdx.x;
    if (i >= fi * fo) return;
    int k = i / fo, c = i % fo;
    WT[(size_t)c * fi + k] = (__bf16)W[i];
}

// Wsp (256 x 256) -> head-reduced Wred (256 x 8, f32, [k*8+h]); bsp -> bred[8]
__global__ void k_wsp_reduce(const float* __restrict__ Wsp, const float* __restrict__ bsp,
                             float* __restrict__ Wred, float* __restrict__ bred) {
    int i = blockIdx.x * 256 + threadIdx.x;
    if (i < 256 * HH) {
        int k = i >> 3, h = i & 7;
        float s = 0.f;
        #pragma unroll
        for (int d = 0; d < HD; ++d) s += Wsp[(size_t)k * DOUT + h * HD + d];
        Wred[k * HH + h] = s;
    }
    if (i < HH) {
        float s = 0.f;
        #pragma unroll
        for (int d = 0; d < HD; ++d) s += bsp[i * HD + d];
        bred[i] = s;
    }
}

// spsum[e,h] = spatial_pos[e,:] . Wred[:,h] + bred[h]   (f32, bandwidth-bound)
// block = 256 threads, 32 edges per block
__global__ void k_spsum(const float* __restrict__ sp, const float* __restrict__ Wred,
                        const float* __restrict__ bred, float* __restrict__ spsum) {
    __shared__ float rows[32 * 256];   // 32 KB
    __shared__ float wl[256 * HH];     // 8 KB
    __shared__ float bl[HH];
    int tid = threadIdx.x;
    size_t e0 = (size_t)blockIdx.x * 32;
    for (int i = tid; i < 32 * 256; i += 256) rows[i] = sp[e0 * 256 + i];
    for (int i = tid; i < 256 * HH; i += 256) wl[i] = Wred[i];
    if (tid < HH) bl[tid] = bred[tid];
    __syncthreads();
    int el = tid >> 3, h = tid & 7;   // 32 edges x 8 heads
    float acc = bl[h];
    const float* r = &rows[el * 256];
    #pragma unroll 8
    for (int k = 0; k < 256; ++k) acc += r[k] * wl[k * HH + h];
    spsum[(e0 + el) * HH + h] = acc;
}

// ---------------------------------------------------------------------------
// Generic bf16 WMMA GEMM: out = A(RxK) @ WT(CxK)^T + bias, f32 accumulate.
// One 16x16 tile per wave, 8 waves per block. R,C multiples of 16; K mult of 32.
#define EPI_F32 0
#define EPI_GELU_BF16 1

template <int EPI>
__global__ void k_gemm_wmma(const __bf16* __restrict__ A, const __bf16* __restrict__ WT,
                            const float* __restrict__ bias,
                            float* __restrict__ outF, __bf16* __restrict__ outB,
                            int R, int Kdim, int C) {
    int wave = threadIdx.x >> 5;
    int colTiles = C >> 4;
    int tile = blockIdx.x * 8 + wave;
    int totalTiles = (R >> 4) * colTiles;
    if (tile >= totalTiles) return;              // whole-wave exit; EXEC stays all-1s
    int rowTile = tile / colTiles;
    int colTile = tile - rowTile * colTiles;
    int lane = threadIdx.x & 31;
    int half = lane >> 4;
    int l15  = lane & 15;

    const __bf16* arow = A  + (size_t)(rowTile * 16 + l15) * Kdim;
    const __bf16* brow = WT + (size_t)(colTile * 16 + l15) * Kdim;

    floatx8 acc = {};
    for (int kk = 0; kk < Kdim; kk += 32) {
        union { uint4 u[2]; bf16x16 v; } fa, fb;
        // A 16x32 bf16 lane layout: elems 0..7 -> K = kk + 8*half + e
        //                           elems 8..15 -> K = kk + 16 + 8*half + e
        fa.u[0] = *(const uint4*)(arow + kk + 8 * half);
        fa.u[1] = *(const uint4*)(arow + kk + 16 + 8 * half);
        // B 32x16 bf16 lane layout: same K mapping as A
        fb.u[0] = *(const uint4*)(brow + kk + 8 * half);
        fb.u[1] = *(const uint4*)(brow + kk + 16 + 8 * half);
        acc = __builtin_amdgcn_wmma_f32_16x16x32_bf16(
            false, fa.v, false, fb.v, (short)0, acc, false, false);
    }

    int nCol = colTile * 16 + l15;
    float bv = bias[nCol];
    #pragma unroll
    for (int r = 0; r < 8; ++r) {
        int m = rowTile * 16 + r + 8 * half;     // C/D layout: VGPR r -> M = r + 8*half
        float val = acc[r] + bv;
        if (EPI == EPI_F32) {
            outF[(size_t)m * C + nCol] = val;
        } else {
            float g = 0.5f * val * (1.f + erff(val * 0.70710678118654752f));
            outB[(size_t)m * C + nCol] = (__bf16)g;
        }
    }
}

// ---------------------------------------------------------------------------
// Edge attention: one wave per edge. Q/K/V f32 (153 MB total) live in 192 MB L2.
__global__ void k_edge(const int* __restrict__ src, const int* __restrict__ dst,
                       const float* __restrict__ Q, const float* __restrict__ K,
                       const float* __restrict__ V, const float* __restrict__ spsum,
                       float* __restrict__ wV, float* __restrict__ z) {
    int e = blockIdx.x * 8 + (threadIdx.x >> 5);
    if (e >= EE) return;
    int lane = threadIdx.x & 31;
    int s = src[e], d = dst[e];
    const float* qrow = Q + (size_t)d * DOUT;
    const float* krow = K + (size_t)s * DOUT;
    const float* vrow = V + (size_t)s * DOUT;
    const float inv_scale = 0.1767766952966369f;   // 1/sqrt(32)
    #pragma unroll
    for (int h = 0; h < HH; ++h) {
        float p = krow[h * HD + lane] * qrow[h * HD + lane];
        #pragma unroll
        for (int m = 16; m >= 1; m >>= 1) p += __shfl_xor(p, m, 32);
        float sc = p * inv_scale + spsum[(size_t)e * HH + h];
        sc = fminf(fmaxf(sc, -5.f), 5.f);
        sc = expf(sc);
        atomicAdd(&wV[(size_t)d * DOUT + h * HD + lane], sc * vrow[h * HD + lane]);
        if (lane == 0) atomicAdd(&z[(size_t)d * HH + h], sc);
    }
}

// h_attn = wV/(z+1e-6); build bf16 concat [h | h_attn] per head
__global__ void k_concat(const float* __restrict__ h, const float* __restrict__ wV,
                         const float* __restrict__ z, float* __restrict__ h_attn,
                         __bf16* __restrict__ concatB) {
    int i = blockIdx.x * 256 + threadIdx.x;   // over N*256
    int n = i >> 8, c = i & 255;
    int hh = c >> 5, dd = c & 31;
    float zz = z[(size_t)n * HH + hh];
    float ha = wV[i] / (zz + 1e-6f);
    h_attn[i] = ha;
    size_t base = (size_t)n * 512 + hh * 64;
    concatB[base + dd]      = (__bf16)h[i];
    concatB[base + 32 + dd] = (__bf16)ha;
}

// xg = sigmoid(gate_pre) * h_attn -> bf16
__global__ void k_gate(const float* __restrict__ gate_pre, const float* __restrict__ h_attn,
                       __bf16* __restrict__ xgB) {
    int i = blockIdx.x * 256 + threadIdx.x;
    float g = 1.f / (1.f + expf(-gate_pre[i]));
    xgB[i] = (__bf16)(g * h_attn[i]);
}

// x = BN_eval(LN(a + resid)); wave per row; optional bf16 copy for next GEMM
template <int WRITE_BF16>
__global__ void k_ln_bn(const float* __restrict__ a, const float* __restrict__ resid,
                        const float* __restrict__ lg, const float* __restrict__ lb,
                        const float* __restrict__ bg, const float* __restrict__ bb,
                        float* __restrict__ outF, __bf16* __restrict__ outB) {
    int row = blockIdx.x * 8 + (threadIdx.x >> 5);
    if (row >= NN) return;
    int lane = threadIdx.x & 31;
    float v[8], s = 0.f, ss = 0.f;
    #pragma unroll
    for (int j = 0; j < 8; ++j) {
        int c = j * 32 + lane;
        float x = a[(size_t)row * DOUT + c] + resid[(size_t)row * DOUT + c];
        v[j] = x; s += x; ss += x * x;
    }
    #pragma unroll
    for (int m = 16; m >= 1; m >>= 1) { s += __shfl_xor(s, m, 32); ss += __shfl_xor(ss, m, 32); }
    float mean = s * (1.f / 256.f);
    float var  = ss * (1.f / 256.f) - mean * mean;
    float inv  = rsqrtf(var + 1e-5f);
    float rbn  = rsqrtf(1.f + 1e-5f);
    #pragma unroll
    for (int j = 0; j < 8; ++j) {
        int c = j * 32 + lane;
        float y = (v[j] - mean) * inv * lg[c] + lb[c];
        y = y * rbn * bg[c] + bb[c];
        outF[(size_t)row * DOUT + c] = y;
        if (WRITE_BF16) outB[(size_t)row * DOUT + c] = (__bf16)y;
    }
}

// ---------------------------------------------------------------------------
extern "C" void kernel_launch(void* const* d_in, const int* in_sizes, int n_in,
                              void* d_out, int out_size, void* d_ws, size_t ws_size,
                              hipStream_t stream) {
    const float* h     = (const float*)d_in[0];
    const float* spos  = (const float*)d_in[1];
    const int*   src   = (const int*)d_in[2];
    const int*   dst   = (const int*)d_in[3];
    const float* Wq = (const float*)d_in[4],  *bq = (const float*)d_in[5];
    const float* Wk = (const float*)d_in[6],  *bk = (const float*)d_in[7];
    const float* Wv = (const float*)d_in[8],  *bv = (const float*)d_in[9];
    const float* Wsp = (const float*)d_in[10], *bsp = (const float*)d_in[11];
    const float* Wo = (const float*)d_in[12], *bo = (const float*)d_in[13];
    const float* Wg = (const float*)d_in[14], *bg = (const float*)d_in[15];
    const float* W1 = (const float*)d_in[16], *b1 = (const float*)d_in[17];
    const float* W2 = (const float*)d_in[18], *b2 = (const float*)d_in[19];
    const float* ln1_g = (const float*)d_in[20], *ln1_b = (const float*)d_in[21];
    const float* ln2_g = (const float*)d_in[22], *ln2_b = (const float*)d_in[23];
    const float* bn1_g = (const float*)d_in[24], *bn1_b = (const float*)d_in[25];
    const float* bn2_g = (const float*)d_in[26], *bn2_b = (const float*)d_in[27];
    float* out = (float*)d_out;

    char* base = (char*)d_ws;
    size_t off = 0;
    auto alloc = [&](size_t bytes) -> char* {
        char* r = base + off;
        off = (off + bytes + 255) & ~(size_t)255;
        return r;
    };
    __bf16* hB      = (__bf16*)alloc((size_t)NN * DIN * 2);
    __bf16* WqT     = (__bf16*)alloc((size_t)DIN * DOUT * 2);
    __bf16* WkT     = (__bf16*)alloc((size_t)DIN * DOUT * 2);
    __bf16* WvT     = (__bf16*)alloc((size_t)DIN * DOUT * 2);
    __bf16* WoT     = (__bf16*)alloc((size_t)DOUT * DOUT * 2);
    __bf16* WgT     = (__bf16*)alloc((size_t)512 * DOUT * 2);
    __bf16* W1T     = (__bf16*)alloc((size_t)DOUT * 4 * DOUT * 2);
    __bf16* W2T     = (__bf16*)alloc((size_t)4 * DOUT * DOUT * 2);
    float*  Wred    = (float*)alloc(256 * HH * 4);
    float*  bred    = (float*)alloc(HH * 4);
    float*  Qb      = (float*)alloc((size_t)NN * DOUT * 4);
    float*  Kb      = (float*)alloc((size_t)NN * DOUT * 4);
    float*  Vb      = (float*)alloc((size_t)NN * DOUT * 4);
    float*  spsum   = (float*)alloc((size_t)EE * HH * 4);
    float*  wVb     = (float*)alloc((size_t)NN * DOUT * 4);
    float*  zb      = (float*)alloc((size_t)NN * HH * 4);
    float*  h_attn  = (float*)alloc((size_t)NN * DOUT * 4);
    __bf16* concatB = (__bf16*)alloc((size_t)NN * 512 * 2);
    float*  gatePre = (float*)alloc((size_t)NN * DOUT * 4);
    __bf16* xgB     = (__bf16*)alloc((size_t)NN * DOUT * 2);
    float*  attnOut = (float*)alloc((size_t)NN * DOUT * 4);
    float*  x1      = (float*)alloc((size_t)NN * DOUT * 4);
    __bf16* x1B     = (__bf16*)alloc((size_t)NN * DOUT * 2);
    __bf16* y1B     = (__bf16*)alloc((size_t)NN * 4 * DOUT * 2);
    float*  y2      = (float*)alloc((size_t)NN * DOUT * 4);
    (void)ws_size; (void)n_in; (void)in_sizes; (void)out_size;

    // zero atomic accumulators (graph-capturable)
    hipMemsetAsync(wVb, 0, (size_t)NN * DOUT * 4, stream);
    hipMemsetAsync(zb,  0, (size_t)NN * HH * 4, stream);

    // prep: conversions + weight transposes + Wsp head-reduction
    k_f32_to_bf16<<<(NN * DIN + 255) / 256, 256, 0, stream>>>(h, hB, NN * DIN);
    k_transpose_bf16<<<(DIN * DOUT + 255) / 256, 256, 0, stream>>>(Wq, WqT, DIN, DOUT);
    k_transpose_bf16<<<(DIN * DOUT + 255) / 256, 256, 0, stream>>>(Wk, WkT, DIN, DOUT);
    k_transpose_bf16<<<(DIN * DOUT + 255) / 256, 256, 0, stream>>>(Wv, WvT, DIN, DOUT);
    k_transpose_bf16<<<(DOUT * DOUT + 255) / 256, 256, 0, stream>>>(Wo, WoT, DOUT, DOUT);
    k_transpose_bf16<<<(512 * DOUT + 255) / 256, 256, 0, stream>>>(Wg, WgT, 512, DOUT);
    k_transpose_bf16<<<(DOUT * 4 * DOUT + 255) / 256, 256, 0, stream>>>(W1, W1T, DOUT, 4 * DOUT);
    k_transpose_bf16<<<(4 * DOUT * DOUT + 255) / 256, 256, 0, stream>>>(W2, W2T, 4 * DOUT, DOUT);
    k_wsp_reduce<<<8, 256, 0, stream>>>(Wsp, bsp, Wred, bred);

    auto gemm_tiles = [](int R, int C) { return (size_t)(R / 16) * (C / 16); };
    // Q, K, V projections (WMMA bf16 -> f32)
    k_gemm_wmma<EPI_F32><<<(gemm_tiles(NN, DOUT) + 7) / 8, 256, 0, stream>>>(
        hB, WqT, bq, Qb, nullptr, NN, DIN, DOUT);
    k_gemm_wmma<EPI_F32><<<(gemm_tiles(NN, DOUT) + 7) / 8, 256, 0, stream>>>(
        hB, WkT, bk, Kb, nullptr, NN, DIN, DOUT);
    k_gemm_wmma<EPI_F32><<<(gemm_tiles(NN, DOUT) + 7) / 8, 256, 0, stream>>>(
        hB, WvT, bv, Vb, nullptr, NN, DIN, DOUT);

    // spatial bias, reduced over head dims (pure bandwidth pass over 819 MB)
    k_spsum<<<EE / 32, 256, 0, stream>>>(spos, Wred, bred, spsum);

    // edge attention with L2-resident gathers + f32 scatter atomics
    k_edge<<<EE / 8, 256, 0, stream>>>(src, dst, Qb, Kb, Vb, spsum, wVb, zb);

    // h_attn + concat, gating GEMM, gate apply, output projection
    k_concat<<<NN, 256, 0, stream>>>(h, wVb, zb, h_attn, concatB);
    k_gemm_wmma<EPI_F32><<<(gemm_tiles(NN, DOUT) + 7) / 8, 256, 0, stream>>>(
        concatB, WgT, bg, gatePre, nullptr, NN, 512, DOUT);
    k_gate<<<NN, 256, 0, stream>>>(gatePre, h_attn, xgB);
    k_gemm_wmma<EPI_F32><<<(gemm_tiles(NN, DOUT) + 7) / 8, 256, 0, stream>>>(
        xgB, WoT, bo, attnOut, nullptr, NN, DOUT, DOUT);

    // residual + LN1 + BN1
    k_ln_bn<1><<<NN / 8, 256, 0, stream>>>(attnOut, h, ln1_g, ln1_b, bn1_g, bn1_b, x1, x1B);

    // FFN: W1 (+exact GELU -> bf16), W2
    k_gemm_wmma<EPI_GELU_BF16><<<(gemm_tiles(NN, 4 * DOUT) + 7) / 8, 256, 0, stream>>>(
        x1B, W1T, b1, nullptr, y1B, NN, DOUT, 4 * DOUT);
    k_gemm_wmma<EPI_F32><<<(gemm_tiles(NN, DOUT) + 7) / 8, 256, 0, stream>>>(
        y1B, W2T, b2, y2, nullptr, NN, 4 * DOUT, DOUT);

    // residual + LN2 + BN2 -> final output
    k_ln_bn<0><<<NN / 8, 256, 0, stream>>>(y2, x1, ln2_g, ln2_b, bn2_g, bn2_b, out, nullptr);
}
